// SingleHeadAttention_47210280518296
// MI455X (gfx1250) — matrix operand order repaired and checked by reading, not compiled, not run on hardware
//
#include <hip/hip_runtime.h>

#define DIMSZ 1024
#define SEQSZ 4096
#define NBATCH 4
#define LN_ROPE_BASE 9.210340371976184f   // ln(10000)
#define SM_SCALE 0.03125f                 // 1/sqrt(1024)
#define V8F_ZERO {0.f,0.f,0.f,0.f,0.f,0.f,0.f,0.f}

typedef __attribute__((ext_vector_type(16))) __bf16 v16bf;
typedef __attribute__((ext_vector_type(8)))  __bf16 v8bf;
typedef __attribute__((ext_vector_type(8)))  float  v8f;

union V16U { v16bf v; v8bf h[2]; };

__device__ __forceinline__ unsigned short f2bf(float f) {
  unsigned int u = __float_as_uint(f);
  u += 0x7FFFu + ((u >> 16) & 1u);
  return (unsigned short)(u >> 16);
}

// ---- CDNA5 async global->LDS copy (ASYNCcnt-tracked), per-lane 16B chunks
__device__ __forceinline__ void async_copy_b128(unsigned lds_off,
                                                const unsigned short* g) {
  asm volatile("global_load_async_to_lds_b128 %0, %1, off"
               :: "v"(lds_off), "v"(g) : "memory");
}
__device__ __forceinline__ void wait_async0() {
  asm volatile("s_wait_asynccnt 0" ::: "memory");
}

// ---------------------------------------------------------------- fp32 -> bf16
__global__ void cvt_f32_bf16_kernel(const float* __restrict__ src,
                                    unsigned short* __restrict__ dst, int n4) {
  int i = blockIdx.x * blockDim.x + threadIdx.x;
  if (i < n4) {
    float4 f = reinterpret_cast<const float4*>(src)[i];
    ushort4 o;
    o.x = f2bf(f.x); o.y = f2bf(f.y); o.z = f2bf(f.z); o.w = f2bf(f.w);
    reinterpret_cast<ushort4*>(dst)[i] = o;
  }
}

// ------------------------------------------------- projections (+RoPE for Q/K)
// C[m,e] = sum_d X[m,d] * W[e,d].  One wave = 16x64 output (4 N-tiles, A reuse).
// mode 0: RoPE, store row-major [B*S, D].  mode 2: no RoPE, store V^T [B, D, S].
__global__ __launch_bounds__(128) void proj_rope_kernel(
    const unsigned short* __restrict__ X, const unsigned short* __restrict__ W,
    unsigned short* __restrict__ Out, int mode) {
  const int lane = threadIdx.x & 31;
  const int wid  = (blockIdx.x << 2) + (threadIdx.x >> 5);
  const int Ngrp = DIMSZ / 64;                       // 16
  const int mt = wid / Ngrp;
  const int ng = wid - mt * Ngrp;
  const int g = lane >> 4, r = lane & 15;
  const unsigned short* aBase = X + (size_t)(mt * 16 + r) * DIMSZ;
  const unsigned short* bBase = W + (size_t)(ng * 64 + r) * DIMSZ;
  v8f acc[4];
#pragma unroll
  for (int nn = 0; nn < 4; ++nn) acc[nn] = (v8f)V8F_ZERO;
#pragma unroll 2
  for (int k0 = 0; k0 < DIMSZ; k0 += 32) {
    V16U A;
    A.h[0] = *reinterpret_cast<const v8bf*>(aBase + k0 + 16 * g);
    A.h[1] = *reinterpret_cast<const v8bf*>(aBase + k0 + 16 * g + 8);
#pragma unroll
    for (int nn = 0; nn < 4; ++nn) {
      V16U B;
      B.v = *reinterpret_cast<const v16bf*>(bBase + (size_t)nn * 16 * DIMSZ +
                                            k0 + 16 * g);
      acc[nn] = __builtin_amdgcn_wmma_f32_16x16x32_bf16(
          false, A.v, false, B.v, (short)0, acc[nn], false, false);
    }
  }
#pragma unroll
  for (int nn = 0; nn < 4; ++nn) {
    const int e = ng * 64 + nn * 16 + r;
    if (mode == 2) {                     // V: transposed store [b, e, s]
#pragma unroll
      for (int j = 0; j < 8; ++j) {
        int srow = mt * 16 + j + 8 * g;
        int b = srow / SEQSZ, s = srow - b * SEQSZ;
        Out[(size_t)(b * DIMSZ + e) * SEQSZ + s] = f2bf(acc[nn][j]);
      }
    } else {                             // Q/K: RoPE then row-major store
      float inv_freq = __expf(-((float)(e & ~1) / (float)DIMSZ) * LN_ROPE_BASE);
#pragma unroll
      for (int j = 0; j < 8; ++j) {
        int srow = mt * 16 + j + 8 * g;
        int pos = srow & (SEQSZ - 1);
        float f = (float)pos * inv_freq;
        float c = __cosf(f), sn = __sinf(f);
        float v = acc[nn][j];
        float p = __shfl_xor(v, 1, 32);  // RoPE partner in adjacent lane
        float o = (e & 1) ? (v * c + p * sn) : (v * c - p * sn);
        Out[(size_t)srow * DIMSZ + e] = f2bf(o);
      }
    }
  }
}

// -------------------------------------------- pass 1: softmax stats (m, l)
__global__ __launch_bounds__(32) void attn_stats_kernel(
    const unsigned short* __restrict__ Q, const unsigned short* __restrict__ K,
    float* __restrict__ Mbuf, float* __restrict__ Lbuf) {
  __shared__ __align__(16) unsigned short qtile[16 * DIMSZ];
  const int lane = threadIdx.x;
  const int qbase = blockIdx.x * 16;
  const int b = qbase / SEQSZ;
  const int pos0 = qbase & (SEQSZ - 1);
  {
    unsigned lds0 = (unsigned)(uintptr_t)&qtile[0];
    const unsigned short* src = Q + (size_t)qbase * DIMSZ;
    for (int idx = lane; idx < 2048; idx += 32)
      async_copy_b128(lds0 + idx * 16, src + idx * 8);
    wait_async0();
  }
  __syncthreads();
  const int g = lane >> 4, r = lane & 15;
  float mrow[8], lrow[8];
#pragma unroll
  for (int j = 0; j < 8; ++j) { mrow[j] = -3.0e38f; lrow[j] = 0.f; }
  const unsigned short* Kb = K + (size_t)b * SEQSZ * DIMSZ;
  const int kvmax = pos0 + 15;
  const int nchunk = kvmax / 32 + 1;
  for (int c = 0; c < nchunk; ++c) {
    const int kv0 = c * 32;
    const unsigned short* bPtr = Kb + (size_t)(kv0 + r) * DIMSZ + 16 * g;
    v8f s0 = V8F_ZERO, s1 = V8F_ZERO;
    for (int k0 = 0; k0 < DIMSZ; k0 += 32) {
      V16U A, B0, B1;
      A.h[0] = *reinterpret_cast<const v8bf*>(qtile + r * DIMSZ + k0 + 16 * g);
      A.h[1] = *reinterpret_cast<const v8bf*>(qtile + r * DIMSZ + k0 + 16 * g + 8);
      B0.v = *reinterpret_cast<const v16bf*>(bPtr + k0);
      B1.v = *reinterpret_cast<const v16bf*>(bPtr + (size_t)16 * DIMSZ + k0);
      s0 = __builtin_amdgcn_wmma_f32_16x16x32_bf16(false, A.v, false, B0.v,
                                                   (short)0, s0, false, false);
      s1 = __builtin_amdgcn_wmma_f32_16x16x32_bf16(false, A.v, false, B1.v,
                                                   (short)0, s1, false, false);
    }
#pragma unroll
    for (int t = 0; t < 2; ++t) {
      const int kvcol = kv0 + t * 16 + r;
      const v8f& s = t ? s1 : s0;
#pragma unroll
      for (int j = 0; j < 8; ++j) {
        if (kvcol <= pos0 + j + 8 * g) {            // causal
          float sv = s[j] * SM_SCALE;
          float mn = fmaxf(mrow[j], sv);
          lrow[j] = lrow[j] * __expf(mrow[j] - mn) + __expf(sv - mn);
          mrow[j] = mn;
        }
      }
    }
  }
#pragma unroll
  for (int off = 1; off <= 8; off <<= 1) {          // merge across 16-lane group
#pragma unroll
    for (int j = 0; j < 8; ++j) {
      float mo = __shfl_xor(mrow[j], off, 32);
      float lo = __shfl_xor(lrow[j], off, 32);
      float mn = fmaxf(mrow[j], mo);
      lrow[j] = lrow[j] * __expf(mrow[j] - mn) + lo * __expf(mo - mn);
      mrow[j] = mn;
    }
  }
  if (r == 0) {
#pragma unroll
    for (int j = 0; j < 8; ++j) {
      int row = qbase + j + 8 * g;
      Mbuf[row] = mrow[j];
      Lbuf[row] = lrow[j];
    }
  }
}

// ------------------------- pass 2: P = exp(S-m)/l, O = P @ V (8 waves/block)
__global__ __launch_bounds__(256) void attn_av_kernel(
    const unsigned short* __restrict__ Q, const unsigned short* __restrict__ K,
    const unsigned short* __restrict__ VT,
    const float* __restrict__ Mbuf, const float* __restrict__ Lbuf,
    float* __restrict__ Out) {
  __shared__ __align__(16) unsigned short qtile[16 * DIMSZ];
  __shared__ __align__(16) unsigned short ptile[8][16 * 32];
  const int tid = threadIdx.x;
  const int lane = tid & 31, w = tid >> 5;
  const int qbase = blockIdx.x * 16;
  const int b = qbase / SEQSZ;
  const int pos0 = qbase & (SEQSZ - 1);
  const int g = lane >> 4, r = lane & 15;
  {
    unsigned lds0 = (unsigned)(uintptr_t)&qtile[0];
    const unsigned short* src = Q + (size_t)qbase * DIMSZ;
    for (int idx = tid; idx < 2048; idx += 256)
      async_copy_b128(lds0 + idx * 16, src + idx * 8);
    wait_async0();
  }
  __syncthreads();
  float mj[8], ilj[8];
#pragma unroll
  for (int j = 0; j < 8; ++j) {
    int row = qbase + j + 8 * g;
    mj[j] = Mbuf[row];
    ilj[j] = 1.0f / Lbuf[row];
  }
  v8f acc[8];
#pragma unroll
  for (int n = 0; n < 8; ++n) acc[n] = (v8f)V8F_ZERO;
  const unsigned short* Kb = K + (size_t)b * SEQSZ * DIMSZ;
  const unsigned short* Vb = VT + (size_t)b * DIMSZ * SEQSZ;
  const int ebase = w * 128;
  const int kvmax = pos0 + 15;
  const int nblk = kvmax / 256 + 1;
  for (int it = 0; it < nblk; ++it) {
    const int kvb = it * 256;
    const int kc = kvb + 32 * w;                    // this wave's score chunk
    if (kc <= kvmax) {
      const unsigned short* bPtr = Kb + (size_t)(kc + r) * DIMSZ + 16 * g;
      v8f s0 = V8F_ZERO, s1 = V8F_ZERO;
      for (int k0 = 0; k0 < DIMSZ; k0 += 32) {
        V16U A, B0, B1;
        A.h[0] = *reinterpret_cast<const v8bf*>(qtile + r * DIMSZ + k0 + 16 * g);
        A.h[1] = *reinterpret_cast<const v8bf*>(qtile + r * DIMSZ + k0 + 16 * g + 8);
        B0.v = *reinterpret_cast<const v16bf*>(bPtr + k0);
        B1.v = *reinterpret_cast<const v16bf*>(bPtr + (size_t)16 * DIMSZ + k0);
        s0 = __builtin_amdgcn_wmma_f32_16x16x32_bf16(false, A.v, false, B0.v,
                                                     (short)0, s0, false, false);
        s1 = __builtin_amdgcn_wmma_f32_16x16x32_bf16(false, A.v, false, B1.v,
                                                     (short)0, s1, false, false);
      }
#pragma unroll
      for (int t = 0; t < 2; ++t) {
        const v8f& s = t ? s1 : s0;
        const int kvcol = kc + t * 16 + r;
#pragma unroll
        for (int j = 0; j < 8; ++j) {
          float p = 0.f;
          if (kvcol <= pos0 + j + 8 * g)
            p = __expf(s[j] * SM_SCALE - mj[j]) * ilj[j];
          ptile[w][(j + 8 * g) * 32 + t * 16 + r] = f2bf(p);
        }
      }
    }
    __syncthreads();
#pragma unroll
    for (int cc = 0; cc < 8; ++cc) {
      const int kvc = kvb + 32 * cc;
      if (kvc <= kvmax) {
        V16U A;
        A.h[0] = *reinterpret_cast<const v8bf*>(&ptile[cc][r * 32 + 16 * g]);
        A.h[1] = *reinterpret_cast<const v8bf*>(&ptile[cc][r * 32 + 16 * g + 8]);
#pragma unroll
        for (int n = 0; n < 8; ++n) {
          V16U B;
          B.v = *reinterpret_cast<const v16bf*>(
              Vb + (size_t)(ebase + n * 16 + r) * SEQSZ + kvc + 16 * g);
          acc[n] = __builtin_amdgcn_wmma_f32_16x16x32_bf16(
              false, A.v, false, B.v, (short)0, acc[n], false, false);
        }
      }
    }
    __syncthreads();
  }
#pragma unroll
  for (int n = 0; n < 8; ++n) {
    const int e = ebase + n * 16 + r;
#pragma unroll
    for (int j = 0; j < 8; ++j)
      Out[(size_t)(qbase + j + 8 * g) * DIMSZ + e] = acc[n][j];
  }
}

// ----------------------------------------------------------------- launcher
extern "C" void kernel_launch(void* const* d_in, const int* in_sizes, int n_in,
                              void* d_out, int out_size, void* d_ws, size_t ws_size,
                              hipStream_t stream) {
  (void)in_sizes; (void)n_in; (void)out_size; (void)ws_size;
  const float* x  = (const float*)d_in[0];
  const float* wq = (const float*)d_in[1];
  const float* wk = (const float*)d_in[2];
  const float* wv = (const float*)d_in[3];
  float* out = (float*)d_out;

  const size_t nX = (size_t)NBATCH * SEQSZ * DIMSZ;   // 16,777,216
  const size_t nW = (size_t)DIMSZ * DIMSZ;            // 1,048,576

  char* p = (char*)d_ws;
  unsigned short* xbf  = (unsigned short*)p; p += nX * 2;
  unsigned short* wqbf = (unsigned short*)p; p += nW * 2;
  unsigned short* wkbf = (unsigned short*)p; p += nW * 2;
  unsigned short* wvbf = (unsigned short*)p; p += nW * 2;
  unsigned short* qbf  = (unsigned short*)p; p += nX * 2;
  unsigned short* kbf  = (unsigned short*)p; p += nX * 2;
  unsigned short* vtbf = (unsigned short*)p; p += nX * 2;
  float* Mbuf = (float*)p; p += (size_t)NBATCH * SEQSZ * 4;
  float* Lbuf = (float*)p;

  // fp32 -> bf16
  cvt_f32_bf16_kernel<<<(int)(nX / 4 / 256), 256, 0, stream>>>(x, xbf, (int)(nX / 4));
  cvt_f32_bf16_kernel<<<(int)(nW / 4 / 256), 256, 0, stream>>>(wq, wqbf, (int)(nW / 4));
  cvt_f32_bf16_kernel<<<(int)(nW / 4 / 256), 256, 0, stream>>>(wk, wkbf, (int)(nW / 4));
  cvt_f32_bf16_kernel<<<(int)(nW / 4 / 256), 256, 0, stream>>>(wv, wvbf, (int)(nW / 4));

  // projections: (B*S/16) M-tiles x (D/64) N-groups, 4 waves per 128-thread block
  const int projBlocks = (int)((nX / DIMSZ / 16) * (DIMSZ / 64) / 4); // 4096
  proj_rope_kernel<<<projBlocks, 128, 0, stream>>>(xbf, wqbf, qbf, 0);
  proj_rope_kernel<<<projBlocks, 128, 0, stream>>>(xbf, wkbf, kbf, 0);
  proj_rope_kernel<<<projBlocks, 128, 0, stream>>>(xbf, wvbf, vtbf, 2);

  const int qtiles = (int)(nX / DIMSZ / 16);          // 1024
  attn_stats_kernel<<<qtiles, 32, 0, stream>>>(qbf, kbf, Mbuf, Lbuf);
  attn_av_kernel<<<qtiles, 256, 0, stream>>>(qbf, kbf, vtbf, Mbuf, Lbuf, out);
}
